// jd_RHGN_39822936768603
// MI455X (gfx1250) — matrix operand order repaired and checked by reading, not compiled, not run on hardware
//
#include <hip/hip_runtime.h>
#include <hip/hip_bf16.h>
#include <stdint.h>

#define NI 100000
#define NU 100000
#define EE 800000
#define HID 128
#define P224 224

typedef __attribute__((ext_vector_type(16))) __bf16 v16bf;
typedef __attribute__((ext_vector_type(8)))  __bf16 v8bf;
typedef __attribute__((ext_vector_type(8)))  float  v8f;

__device__ __forceinline__ unsigned short bfbits(float f) {
  unsigned u = __float_as_uint(f);
  u += 0x7fffu + ((u >> 16) & 1u);          // round-to-nearest-even
  return (unsigned short)(u >> 16);
}
__device__ __forceinline__ float geluf(float x) {
  return 0.5f * x * (1.0f + erff(x * 0.70710678118654752f));
}
// order-preserving float<->uint encode for atomicMax-based segment max
__device__ __forceinline__ unsigned fenc(float f) {
  unsigned u = __float_as_uint(f);
  return (u & 0x80000000u) ? ~u : (u | 0x80000000u);
}
__device__ __forceinline__ float fdec(unsigned u) {
  return __uint_as_float((u & 0x80000000u) ? (u & 0x7fffffffu) : ~u);
}
// guarantee native GLOBAL_ATOMIC_ADD_F32 (no CAS loop)
__device__ __forceinline__ void fatomic_add(float* p, float v) { unsafeAtomicAdd(p, v); }

// ---------------------------------------------------------------------------
// bf16 WMMA GEMM:  Y[M, ncols] = A[M, lda] * WT^T (+ bias) with epilogues.
//   A  : bf16, row-major, lda multiple of 32 (zero padded)
//   WT : bf16, [ntiles*16][lda]  (row o = output column o of W, padded)
//   8 waves / 256-thread workgroup; wave w handles output col-tiles w, w+8,...
//   grid.x = M/16 (M is a multiple of 16 here: 100000 = 6250*16)
// EPI: 0 = plain, 1 = GELU, 2 = item skip-fusion (alpha*0.25*v + (1-a)*feat)
// ---------------------------------------------------------------------------
template<int EPI, bool WF32, bool WBF16>
__global__ __launch_bounds__(256)
void wmma_gemm(const unsigned short* __restrict__ A,
               const unsigned short* __restrict__ WT,
               const float* __restrict__ bias,
               float* __restrict__ Yf,
               unsigned short* __restrict__ Ybf,
               int lda, int ntiles, int ldy,
               const float* __restrict__ skipw,
               const float* __restrict__ feat, int featld, int featcols)
{
  const int lane = threadIdx.x & 31;
  const int wv   = threadIdx.x >> 5;
  const int hl   = lane >> 4;     // lane-half: K sub-chunk selector
  const int l15  = lane & 15;     // A-row within tile / B-column within tile
  const size_t mbase = (size_t)blockIdx.x * 16;
  const unsigned short* arow = A + (mbase + (size_t)l15) * (size_t)lda;

  for (int nt = wv; nt < ntiles; nt += 8) {
    const unsigned short* wrow = WT + (size_t)(nt * 16 + l15) * (size_t)lda;
    v8f acc = {};
    for (int kb = 0; kb < lda; kb += 32) {
      // A 16x32 bf16 fragment: lane-half h holds K = kb+16h..kb+16h+15
      // of row M=l15 (contiguous along K, matching B)
      v8bf a0 = *(const v8bf*)(arow + kb + hl * 16);
      v8bf a1 = *(const v8bf*)(arow + kb + hl * 16 + 8);
      // B 32x16 bf16 fragment: lane-half h holds K = kb+16h..kb+16h+15 of
      // column N=l15 (contiguous along K in the pre-transposed weight)
      v16bf a, b;
#pragma unroll
      for (int i = 0; i < 8; ++i) { a[i] = a0[i]; a[i + 8] = a1[i]; }
      b = *(const v16bf*)(wrow + kb + hl * 16);
      acc = __builtin_amdgcn_wmma_f32_16x16x32_bf16(false, a, false, b,
                                                    (short)0, acc, false, false);
    }
    const int col = nt * 16 + l15;
    const float bc = bias ? bias[col] : 0.0f;
    float alpha = 0.0f;
    if (EPI == 2) alpha = 1.0f / (1.0f + __expf(-skipw[0]));
#pragma unroll
    for (int r = 0; r < 8; ++r) {              // C/D: VGPR r -> M = r + 8*half
      const size_t row = mbase + (size_t)(hl * 8 + r);
      float v = acc[r] + bc;
      if (EPI == 1) v = geluf(v);
      if (EPI == 2) {
        const float f = (col < featcols) ? feat[row * (size_t)featld + col] : 0.0f;
        v = alpha * 0.25f * v + (1.0f - alpha) * f;   // uniform 4-way softmax
      }
      if (WF32)  Yf [row * (size_t)ldy + col] = v;
      if (WBF16) Ybf[row * (size_t)ldy + col] = bfbits(v);
    }
  }
}

// ---------------------------------------------------------------------------
// Weight prep: fp32 W[din,dout] (optionally fused with per-head relation
// transform R[8,16,16], i.e. W' = W * blockdiag(R)) -> bf16 W^T padded
// [doutp][dinp], plus fused/padded fp32 bias.
// ---------------------------------------------------------------------------
__global__ void prep_wT(const float* __restrict__ W, const float* __restrict__ B,
                        const float* __restrict__ R,
                        int din, int dout, int dinp, int doutp,
                        unsigned short* __restrict__ WTo, float* __restrict__ bo)
{
  int t = blockIdx.x * blockDim.x + threadIdx.x;
  if (t >= dinp * doutp) return;
  int o = t / dinp, i = t % dinp;
  float acc = 0.0f;
  if (i < din && o < dout) {
    if (R) {
      int h = o >> 4, e = o & 15;
#pragma unroll
      for (int d = 0; d < 16; ++d)
        acc += W[(size_t)i * dout + h * 16 + d] * R[(h * 16 + d) * 16 + e];
    } else acc = W[(size_t)i * dout + o];
  }
  WTo[(size_t)o * dinp + i] = bfbits(acc);
  if (i == 0) {
    float bb = 0.0f;
    if (o < dout) {
      if (R) {
        int h = o >> 4, e = o & 15;
#pragma unroll
        for (int d = 0; d < 16; ++d) bb += B[h * 16 + d] * R[(h * 16 + d) * 16 + e];
      } else bb = B[o];
    }
    bo[o] = bb;
  }
}

// cid-gather + fp32 -> padded bf16
__global__ void gather_cid_bf16(const float* __restrict__ tab, const int* __restrict__ cid,
                                unsigned short* __restrict__ out, int n)
{
  int t = blockIdx.x * blockDim.x + threadIdx.x;
  if (t >= n) return;
  int row = t / P224, c = t % P224;
  float v = (c < 200) ? tab[(size_t)cid[row] * 200 + c] : 0.0f;
  out[t] = bfbits(v);
}
__global__ void conv_feat_bf16(const float* __restrict__ in, unsigned short* __restrict__ out,
                               int n, int incols, int outcols)
{
  int t = blockIdx.x * blockDim.x + threadIdx.x;
  if (t >= n) return;
  int row = t / outcols, c = t % outcols;
  float v = (c < incols) ? in[(size_t)row * incols + c] : 0.0f;
  out[t] = bfbits(v);
}
__global__ void gelu_to_bf16(const float* __restrict__ in, unsigned short* __restrict__ out, int n)
{
  int t = blockIdx.x * blockDim.x + threadIdx.x;
  if (t >= n) return;
  out[t] = bfbits(geluf(in[t]));
}

// ---------------------------------------------------------------------------
// Edge softmax-aggregate, 3 passes over (edge, head) pairs.
// K'/V' already carry the relation transform (folded into weights).
// ---------------------------------------------------------------------------
__global__ void edge_score(const float* __restrict__ Kp, const float* __restrict__ Q,
                           const float* __restrict__ pri,
                           const int* __restrict__ src, const int* __restrict__ dst,
                           int ne, float* __restrict__ sc, unsigned* __restrict__ mx)
{
  int t = blockIdx.x * blockDim.x + threadIdx.x;
  if (t >= ne * 8) return;
  int e = t >> 3, h = t & 7;
  int s = src[e], d = dst[e];
  const float4* kp = (const float4*)(Kp + ((size_t)s * 8 + h) * 16);
  const float4* qp = (const float4*)(Q  + ((size_t)d * 8 + h) * 16);
  float acc = 0.0f;
#pragma unroll
  for (int i = 0; i < 4; ++i) {
    float4 k4 = kp[i], q4 = qp[i];
    acc += k4.x * q4.x + k4.y * q4.y + k4.z * q4.z + k4.w * q4.w;
  }
  acc *= pri[h] * 0.25f;                    // rel_pri / sqrt(DK=16)
  sc[t] = acc;
  atomicMax(mx + (size_t)d * 8 + h, fenc(acc));
}
__global__ void edge_norm(const int* __restrict__ dst, int ne, float* __restrict__ sc,
                          const unsigned* __restrict__ mx, float* __restrict__ ssum)
{
  int t = blockIdx.x * blockDim.x + threadIdx.x;
  if (t >= ne * 8) return;
  int e = t >> 3, h = t & 7;
  int d = dst[e];
  float ex = __expf(sc[t] - fdec(mx[(size_t)d * 8 + h]));
  sc[t] = ex;
  fatomic_add(ssum + (size_t)d * 8 + h, ex);
}
__global__ void edge_agg(const float* __restrict__ Vp, const int* __restrict__ src,
                         const int* __restrict__ dst, int ne,
                         const float* __restrict__ sc, const float* __restrict__ ssum,
                         float* __restrict__ agg)
{
  int t = blockIdx.x * blockDim.x + threadIdx.x;
  if (t >= ne * 8) return;
  int e = t >> 3, h = t & 7;
  int s = src[e], d = dst[e];
  float att = sc[t] / ssum[(size_t)d * 8 + h];
  const float4* vp = (const float4*)(Vp + ((size_t)s * 8 + h) * 16);
  float* ag = agg + ((size_t)d * 8 + h) * 16;
#pragma unroll
  for (int i = 0; i < 4; ++i) {
    float4 v = vp[i];
    fatomic_add(ag + 4 * i + 0, att * v.x);
    fatomic_add(ag + 4 * i + 1, att * v.y);
    fatomic_add(ag + 4 * i + 2, att * v.z);
    fatomic_add(ag + 4 * i + 3, att * v.w);
  }
}

// skip connection + LayerNorm over 128 cols; one row per 128-thread block
__global__ __launch_bounds__(128)
void skip_ln(const float* __restrict__ trans, const float* __restrict__ hprev,
             const float* __restrict__ skipw, int nid,
             const float* __restrict__ g, const float* __restrict__ b,
             float* __restrict__ hf, unsigned short* __restrict__ hbf)
{
  __shared__ float sm[128];
  const int row = blockIdx.x, c = threadIdx.x;
  const size_t idx = (size_t)row * HID + c;
  const float alpha = 1.0f / (1.0f + __expf(-skipw[nid]));
  const float x = alpha * trans[idx] + (1.0f - alpha) * hprev[idx];
  sm[c] = x; __syncthreads();
  for (int s = 64; s > 0; s >>= 1) { if (c < s) sm[c] += sm[c + s]; __syncthreads(); }
  const float mu = sm[0] / 128.0f; __syncthreads();
  const float dx = x - mu;
  sm[c] = dx * dx; __syncthreads();
  for (int s = 64; s > 0; s >>= 1) { if (c < s) sm[c] += sm[c + s]; __syncthreads(); }
  const float var = sm[0] / 128.0f;
  const float y = dx * rsqrtf(var + 1e-5f) * g[c] + b[c];
  hf[idx] = y;
  hbf[idx] = bfbits(y);
}

__global__ void copy_labels(const int* __restrict__ lab, float* __restrict__ out, int n)
{
  int t = blockIdx.x * blockDim.x + threadIdx.x;
  if (t < n) out[t] = (float)lab[t];
}

// ---------------------------------------------------------------------------
extern "C" void kernel_launch(void* const* d_in, const int* in_sizes, int n_in,
                              void* d_out, int out_size, void* d_ws, size_t ws_size,
                              hipStream_t stream)
{
  (void)in_sizes; (void)n_in; (void)out_size; (void)ws_size;
  const float* item_feat = (const float*)d_in[0];
  const float* user_feat = (const float*)d_in[1];
  const float* cid_table = (const float*)d_in[2];
  const float* P         = (const float*)d_in[3];
  const int* item_cid1 = (const int*)d_in[4];
  const int* src_iu = (const int*)d_in[5];
  const int* dst_iu = (const int*)d_in[6];
  const int* src_ui = (const int*)d_in[7];
  const int* dst_ui = (const int*)d_in[8];
  const int* labels = (const int*)d_in[9];
  float* out = (float*)d_out;

  // ---- params flat offsets (jax pytree = sorted dict keys) ----
  // top: adapt, key, layers, out, query, skip, value
  size_t off = 0;
  auto take = [&](size_t n) { size_t o = off; off += n; return o; };
  size_t ad0W = take(25600), ad0b = take(128), ad1W = take(25600), ad1b = take(128);
  take(40000); take(200);   // key.W / key.b (cancel: uniform softmax)
  struct LOff {
    size_t aW[2], ab[2], kW[2], kb_[2], lnb[2], lng[2], qW[2], qb[2];
    size_t relA, relM, relP, skip, vW[2], vb[2];
  } L[2];
  for (int l = 0; l < 2; ++l) {
    L[l].aW[0] = take(16384); L[l].ab[0] = take(128);
    L[l].aW[1] = take(16384); L[l].ab[1] = take(128);
    L[l].kW[0] = take(16384); L[l].kb_[0] = take(128);
    L[l].kW[1] = take(16384); L[l].kb_[1] = take(128);
    L[l].lnb[0] = take(128); L[l].lng[0] = take(128);
    L[l].lnb[1] = take(128); L[l].lng[1] = take(128);
    L[l].qW[0] = take(16384); L[l].qb[0] = take(128);
    L[l].qW[1] = take(16384); L[l].qb[1] = take(128);
    L[l].relA = take(4096); L[l].relM = take(4096);
    L[l].relP = take(16);   L[l].skip = take(2);
    L[l].vW[0] = take(16384); L[l].vb[0] = take(128);
    L[l].vW[1] = take(16384); L[l].vb[1] = take(128);
  }
  size_t outW = take(2048), outb = take(16);
  take(40000); take(200);   // query (cancels)
  size_t skip0 = take(1);
  size_t valW = take(40000), valb = take(200);

  // ---- workspace bump allocator ----
  uintptr_t cur = (uintptr_t)d_ws;
  auto alloc = [&](size_t bytes) -> void* {
    cur = (cur + 255) & ~(uintptr_t)255;
    void* p = (void*)cur; cur += bytes; return p;
  };
  unsigned short* c_bf  = (unsigned short*)alloc((size_t)NI * P224 * 2);
  unsigned short* uf_bf = (unsigned short*)alloc((size_t)NU * P224 * 2);
  unsigned short* if_bf = (unsigned short*)alloc((size_t)NI * P224 * 2);
  float* hf[2];  unsigned short* hbf[2];
  float *Kp[2], *Qm[2], *Vp[2], *agg[2], *trans[2], *ssum[2];
  unsigned short* gbf[2];
  unsigned* mx[2];
  const int NN[2] = { NI, NU };
  for (int t = 0; t < 2; ++t) {
    hf[t]    = (float*)alloc((size_t)NN[t] * HID * 4);
    hbf[t]   = (unsigned short*)alloc((size_t)NN[t] * HID * 2);
    Kp[t]    = (float*)alloc((size_t)NN[t] * HID * 4);
    Qm[t]    = (float*)alloc((size_t)NN[t] * HID * 4);
    Vp[t]    = (float*)alloc((size_t)NN[t] * HID * 4);
    agg[t]   = (float*)alloc((size_t)NN[t] * HID * 4);
    trans[t] = (float*)alloc((size_t)NN[t] * HID * 4);
    gbf[t]   = (unsigned short*)alloc((size_t)NN[t] * HID * 2);
    mx[t]    = (unsigned*)alloc((size_t)NN[t] * 8 * 4);
    ssum[t]  = (float*)alloc((size_t)NN[t] * 8 * 4);
  }
  float* sc[2];
  sc[0] = (float*)alloc((size_t)EE * 8 * 4);
  sc[1] = (float*)alloc((size_t)EE * 8 * 4);
  // prepped weights (bf16 transposed, padded) + biases
  unsigned short* WvT = (unsigned short*)alloc((size_t)P224 * P224 * 2);
  float* bv224 = (float*)alloc(P224 * 4);
  unsigned short* adT[2]; float* adb[2];
  for (int t = 0; t < 2; ++t) {
    adT[t] = (unsigned short*)alloc((size_t)HID * P224 * 2);
    adb[t] = (float*)alloc(HID * 4);
  }
  unsigned short *kT[2][2], *qT[2][2], *vT[2][2], *aT[2][2];
  float *kb2[2][2], *qb2[2][2], *vb2[2][2], *ab2[2][2];
  for (int l = 0; l < 2; ++l)
    for (int t = 0; t < 2; ++t) {
      kT[l][t] = (unsigned short*)alloc((size_t)HID * HID * 2); kb2[l][t] = (float*)alloc(HID * 4);
      qT[l][t] = (unsigned short*)alloc((size_t)HID * HID * 2); qb2[l][t] = (float*)alloc(HID * 4);
      vT[l][t] = (unsigned short*)alloc((size_t)HID * HID * 2); vb2[l][t] = (float*)alloc(HID * 4);
      aT[l][t] = (unsigned short*)alloc((size_t)HID * HID * 2); ab2[l][t] = (float*)alloc(HID * 4);
    }
  unsigned short* outT = (unsigned short*)alloc((size_t)16 * HID * 2);
  float* outb2 = (float*)alloc(16 * 4);

  const int TPB = 256;
  auto cdiv = [](long long a, long long b) { return (unsigned)((a + b - 1) / b); };

  // ---- 1. weight prep ----
  prep_wT<<<cdiv((size_t)P224 * P224, TPB), TPB, 0, stream>>>(
      P + valW, P + valb, nullptr, 200, 200, P224, P224, WvT, bv224);
  prep_wT<<<cdiv((size_t)HID * P224, TPB), TPB, 0, stream>>>(
      P + ad0W, P + ad0b, nullptr, 200, HID, P224, HID, adT[0], adb[0]);
  prep_wT<<<cdiv((size_t)HID * P224, TPB), TPB, 0, stream>>>(
      P + ad1W, P + ad1b, nullptr, 200, HID, P224, HID, adT[1], adb[1]);
  for (int l = 0; l < 2; ++l)
    for (int t = 0; t < 2; ++t) {
      const unsigned g128 = cdiv((size_t)HID * HID, TPB);
      prep_wT<<<g128, TPB, 0, stream>>>(P + L[l].kW[t], P + L[l].kb_[t],
          P + L[l].relA + (size_t)t * 2048, HID, HID, HID, HID, kT[l][t], kb2[l][t]);
      prep_wT<<<g128, TPB, 0, stream>>>(P + L[l].qW[t], P + L[l].qb[t],
          nullptr, HID, HID, HID, HID, qT[l][t], qb2[l][t]);
      prep_wT<<<g128, TPB, 0, stream>>>(P + L[l].vW[t], P + L[l].vb[t],
          P + L[l].relM + (size_t)t * 2048, HID, HID, HID, HID, vT[l][t], vb2[l][t]);
      prep_wT<<<g128, TPB, 0, stream>>>(P + L[l].aW[t], P + L[l].ab[t],
          nullptr, HID, HID, HID, HID, aT[l][t], ab2[l][t]);
    }
  prep_wT<<<cdiv((size_t)16 * HID, TPB), TPB, 0, stream>>>(
      P + outW, P + outb, nullptr, HID, 16, HID, 16, outT, outb2);

  // ---- 2. inputs -> bf16 ----
  gather_cid_bf16<<<cdiv((size_t)NI * P224, TPB), TPB, 0, stream>>>(
      cid_table, item_cid1, c_bf, NI * P224);
  conv_feat_bf16<<<cdiv((size_t)NU * P224, TPB), TPB, 0, stream>>>(
      user_feat, uf_bf, NU * P224, 200, P224);

  // ---- 3. item category fusion: item_f = a*0.25*(c@Wv+bv) + (1-a)*item_feat ----
  wmma_gemm<2, false, true><<<NI / 16, TPB, 0, stream>>>(
      c_bf, WvT, bv224, nullptr, if_bf, P224, 14, P224,
      P + skip0, item_feat, 200, 200);

  // ---- 4. adapt (GELU) ----
  wmma_gemm<1, true, true><<<NI / 16, TPB, 0, stream>>>(
      if_bf, adT[0], adb[0], hf[0], hbf[0], P224, 8, HID, nullptr, nullptr, 0, 0);
  wmma_gemm<1, true, true><<<NU / 16, TPB, 0, stream>>>(
      uf_bf, adT[1], adb[1], hf[1], hbf[1], P224, 8, HID, nullptr, nullptr, 0, 0);

  // ---- 5. HGT layers ----
  for (int l = 0; l < 2; ++l) {
    for (int t = 0; t < 2; ++t) {
      wmma_gemm<0, true, false><<<NN[t] / 16, TPB, 0, stream>>>(
          hbf[t], kT[l][t], kb2[l][t], Kp[t], nullptr, HID, 8, HID, nullptr, nullptr, 0, 0);
      wmma_gemm<0, true, false><<<NN[t] / 16, TPB, 0, stream>>>(
          hbf[t], qT[l][t], qb2[l][t], Qm[t], nullptr, HID, 8, HID, nullptr, nullptr, 0, 0);
      wmma_gemm<0, true, false><<<NN[t] / 16, TPB, 0, stream>>>(
          hbf[t], vT[l][t], vb2[l][t], Vp[t], nullptr, HID, 8, HID, nullptr, nullptr, 0, 0);
      hipMemsetAsync(mx[t],   0, (size_t)NN[t] * 8 * 4, stream);
      hipMemsetAsync(ssum[t], 0, (size_t)NN[t] * 8 * 4, stream);
      hipMemsetAsync(agg[t],  0, (size_t)NN[t] * HID * 4, stream);
    }
    const unsigned ge = cdiv((size_t)EE * 8, TPB);
    // eid 0: item -> user
    edge_score<<<ge, TPB, 0, stream>>>(Kp[0], Qm[1], P + L[l].relP + 0,
                                       src_iu, dst_iu, EE, sc[0], mx[1]);
    // eid 1: user -> item
    edge_score<<<ge, TPB, 0, stream>>>(Kp[1], Qm[0], P + L[l].relP + 8,
                                       src_ui, dst_ui, EE, sc[1], mx[0]);
    edge_norm<<<ge, TPB, 0, stream>>>(dst_iu, EE, sc[0], mx[1], ssum[1]);
    edge_norm<<<ge, TPB, 0, stream>>>(dst_ui, EE, sc[1], mx[0], ssum[0]);
    edge_agg<<<ge, TPB, 0, stream>>>(Vp[0], src_iu, dst_iu, EE, sc[0], ssum[1], agg[1]);
    edge_agg<<<ge, TPB, 0, stream>>>(Vp[1], src_ui, dst_ui, EE, sc[1], ssum[0], agg[0]);
    for (int t = 0; t < 2; ++t) {
      gelu_to_bf16<<<cdiv((size_t)NN[t] * HID, TPB), TPB, 0, stream>>>(
          agg[t], gbf[t], NN[t] * HID);
      wmma_gemm<0, true, false><<<NN[t] / 16, TPB, 0, stream>>>(
          gbf[t], aT[l][t], ab2[l][t], trans[t], nullptr, HID, 8, HID, nullptr, nullptr, 0, 0);
      skip_ln<<<NN[t], 128, 0, stream>>>(trans[t], hf[t], P + L[l].skip, t,
                                         P + L[l].lng[t], P + L[l].lnb[t],
                                         hf[t], hbf[t]);
    }
  }

  // ---- 6. head + labels ----
  wmma_gemm<0, true, false><<<NU / 16, TPB, 0, stream>>>(
      hbf[1], outT, outb2, out, nullptr, HID, 1, 16, nullptr, nullptr, 0, 0);
  copy_labels<<<cdiv(NU, TPB), TPB, 0, stream>>>(labels, out + (size_t)NU * 16, NU);
}
